// SpectralLayer_22247930593385
// MI455X (gfx1250) — matrix operand order repaired and checked by reading, not compiled, not run on hardware
//
#include <hip/hip_runtime.h>

#define DIM   1024
#define BATCH 32768

typedef __attribute__((ext_vector_type(8)))  float  v8f;
typedef __attribute__((ext_vector_type(16))) __bf16 v16bf;
typedef __attribute__((ext_vector_type(8)))  __bf16 bf16x8;
typedef __attribute__((ext_vector_type(4)))  int    v4i;

#if __has_builtin(__builtin_amdgcn_global_load_async_to_lds_b128) && \
    __has_builtin(__builtin_amdgcn_s_wait_asynccnt)
#define USE_ASYNC 1
#else
#define USE_ASYNC 0
#endif

// ---- f32 -> bf16 helpers ----
__device__ __forceinline__ unsigned short bf16_rne(float f) {
  unsigned u = __float_as_uint(f);
  u += 0x7FFFu + ((u >> 16) & 1u);
  return (unsigned short)(u >> 16);
}
// hi = truncated top 16 bits (exact residual), lo = RNE(residual).
// hi*hi + hi*lo + lo*hi keeps ~2^-16 relative error like full bf16x3.
__device__ __forceinline__ void split_bf16_fast(float f, unsigned short& hi, unsigned short& lo) {
  unsigned u = __float_as_uint(f);
  hi = (unsigned short)(u >> 16);
  float fh = __uint_as_float(u & 0xFFFF0000u);
  lo = bf16_rne(f - fh);
}

#if USE_ASYNC
// global_load_async_to_lds_b128: (AS1 int4* src, AS3 int4* lds_dst, imm offset, imm cpol)
__device__ __forceinline__ void async_cp16(const void* g, void* l) {
  __builtin_amdgcn_global_load_async_to_lds_b128(
      (__attribute__((address_space(1))) v4i*)(g),
      (__attribute__((address_space(3))) v4i*)(l),
      0, 0);
}
#endif

// ---- elementwise prep: layer, (2I - layer), diag ----
__global__ __launch_bounds__(256) void prep_kernel(
    const float* __restrict__ base, const float* __restrict__ base_mask,
    const float* __restrict__ eye,
    const float* __restrict__ tdiag, const float* __restrict__ tdiag_mask,
    const float* __restrict__ udiag,
    float* __restrict__ layer, float* __restrict__ neg2, float* __restrict__ diagM) {
  int i = blockIdx.x * 256 + threadIdx.x;
  float e = eye[i];
  float l = fmaf(base[i], base_mask[i], e);
  layer[i] = l;
  neg2[i]  = 2.0f * e - l;
  diagM[i] = fmaf(tdiag[i], tdiag_mask[i], udiag[i]);
}

// ---- split W (f32 [n][k]) into Whi/Wlo bf16 once ----
__global__ __launch_bounds__(256) void splitw_kernel(
    const float* __restrict__ W,
    unsigned short* __restrict__ Whi, unsigned short* __restrict__ Wlo) {
  int i = blockIdx.x * 256 + threadIdx.x;
  unsigned short h, l;
  split_bf16_fast(W[i], h, l);
  Whi[i] = h;
  Wlo[i] = l;
}

// =====================================================================
// Small bf16x3 WMMA GEMM: C[m,n] = sum_k A[m,k] * B[k,n]   (B is [K][N])
// Tile 128x64, BK=32, 8 waves. Used only for the two 1024^3 prep GEMMs.
// =====================================================================
__global__ __launch_bounds__(256) void gemm_small(
    const float* __restrict__ A, const float* __restrict__ B, float* __restrict__ C,
    int K, int lda, int ldb, int ldc) {
  __shared__ __attribute__((aligned(32))) unsigned short Ahi[128 * 32];
  __shared__ __attribute__((aligned(32))) unsigned short Alo[128 * 32];
  __shared__ __attribute__((aligned(32))) unsigned short Bhi[64 * 32];
  __shared__ __attribute__((aligned(32))) unsigned short Blo[64 * 32];

  const int tid  = threadIdx.x;
  const int lane = tid & 31, wave = tid >> 5;
  const int wm = wave & 3, wn = wave >> 2;
  const int half = lane >> 4, l16 = lane & 15;
  const int mBase = blockIdx.y * 128;
  const int nBase = blockIdx.x * 64;

  v8f acc[2][2] = {};

  for (int k0 = 0; k0 < K; k0 += 32) {
#pragma unroll
    for (int r = 0; r < 4; ++r) {
      int idx = tid + r * 256;
      int row = idx >> 3, c4 = (idx & 7) << 2;
      float4 v = *(const float4*)(A + (size_t)(mBase + row) * lda + k0 + c4);
      float vv[4] = {v.x, v.y, v.z, v.w};
#pragma unroll
      for (int j = 0; j < 4; ++j) {
        unsigned short h, l;
        split_bf16_fast(vv[j], h, l);
        Ahi[row * 32 + c4 + j] = h;
        Alo[row * 32 + c4 + j] = l;
      }
    }
#pragma unroll
    for (int r = 0; r < 2; ++r) {
      int idx = tid + r * 256;
      int kr = idx >> 4, n4 = (idx & 15) << 2;
      float4 v = *(const float4*)(B + (size_t)(k0 + kr) * ldb + nBase + n4);
      float vv[4] = {v.x, v.y, v.z, v.w};
#pragma unroll
      for (int j = 0; j < 4; ++j) {
        unsigned short h, l;
        split_bf16_fast(vv[j], h, l);
        Bhi[(n4 + j) * 32 + kr] = h;
        Blo[(n4 + j) * 32 + kr] = l;
      }
    }
    __syncthreads();

    v16bf ah[2], al[2], bh[2], bl[2];
#pragma unroll
    for (int i = 0; i < 2; ++i) {
      int m = wm * 32 + i * 16 + l16;
      bf16x8 h0 = *(const bf16x8*)(Ahi + m * 32 + 16 * half);
      bf16x8 h1 = *(const bf16x8*)(Ahi + m * 32 + 16 * half + 8);
      bf16x8 l0 = *(const bf16x8*)(Alo + m * 32 + 16 * half);
      bf16x8 l1 = *(const bf16x8*)(Alo + m * 32 + 16 * half + 8);
      ah[i] = __builtin_shufflevector(h0, h1, 0,1,2,3,4,5,6,7,8,9,10,11,12,13,14,15);
      al[i] = __builtin_shufflevector(l0, l1, 0,1,2,3,4,5,6,7,8,9,10,11,12,13,14,15);
    }
#pragma unroll
    for (int j = 0; j < 2; ++j) {
      int n = wn * 32 + j * 16 + l16;
      bh[j] = *(const v16bf*)(Bhi + n * 32 + 16 * half);
      bl[j] = *(const v16bf*)(Blo + n * 32 + 16 * half);
    }
#pragma unroll
    for (int i = 0; i < 2; ++i) {
#pragma unroll
      for (int j = 0; j < 2; ++j) {
        acc[i][j] = __builtin_amdgcn_wmma_f32_16x16x32_bf16(
            false, ah[i], false, bh[j], (short)0, acc[i][j], false, false);
        acc[i][j] = __builtin_amdgcn_wmma_f32_16x16x32_bf16(
            false, ah[i], false, bl[j], (short)0, acc[i][j], false, false);
        acc[i][j] = __builtin_amdgcn_wmma_f32_16x16x32_bf16(
            false, al[i], false, bh[j], (short)0, acc[i][j], false, false);
      }
    }
    __syncthreads();
  }

#pragma unroll
  for (int i = 0; i < 2; ++i) {
#pragma unroll
    for (int j = 0; j < 2; ++j) {
      int nCol = nBase + wn * 32 + j * 16 + l16;
#pragma unroll
      for (int r = 0; r < 8; ++r) {
        int mRow = mBase + wm * 32 + i * 16 + half * 8 + r;
        C[(size_t)mRow * ldc + nCol] = acc[i][j][r];
      }
    }
  }
}

// =====================================================================
// Main GEMM: out[b,d] = eps( sum_k data[b,k] * W[d,k] ), W pre-split bf16.
// Tile 128(M) x 64(N), BK=32. blockIdx.x = N (fastest) for L2 reuse of A.
// Double-buffered LDS staging with async global->LDS copies when available.
// =====================================================================
__global__ __launch_bounds__(256) void main_gemm(
    const float* __restrict__ A,                 // data [BATCH][DIM]
    const unsigned short* __restrict__ BHi,      // Whi  [DIM][DIM] ([n][k])
    const unsigned short* __restrict__ BLo,      // Wlo  [DIM][DIM]
    float* __restrict__ C,                       // out  [BATCH][DIM]
    const float* __restrict__ mask1, const float* __restrict__ mask2) {
  const int K = DIM, LDA = DIM, LDB = DIM, LDC = DIM;
  __shared__ __attribute__((aligned(16))) float          Araw[2][128 * 32];
  __shared__ __attribute__((aligned(32))) unsigned short Ahi[128 * 32];
  __shared__ __attribute__((aligned(32))) unsigned short Alo[128 * 32];
  __shared__ __attribute__((aligned(32))) unsigned short Bhs[2][64 * 32];
  __shared__ __attribute__((aligned(32))) unsigned short Bls[2][64 * 32];

  const int tid  = threadIdx.x;
  const int lane = tid & 31, wave = tid >> 5;
  const int wm = wave & 3, wn = wave >> 2;
  const int half = lane >> 4, l16 = lane & 15;
  const int nBase = blockIdx.x * 64;    // N varies fastest across the grid
  const int mBase = blockIdx.y * 128;

  auto stage = [&](int buf, int k0) {
    // A tile: 128x32 f32 = 1024 x 16B chunks, 4 per thread
#pragma unroll
    for (int r = 0; r < 4; ++r) {
      int idx = tid + r * 256;
      int row = idx >> 3;
      int c   = (idx & 7) << 2;
      const float* g = A + (size_t)(mBase + row) * LDA + k0 + c;
      float* l = &Araw[buf][row * 32 + c];
#if USE_ASYNC
      async_cp16(g, l);
#else
      *(float4*)l = *(const float4*)g;
#endif
    }
    // B tiles: 64x32 u16 = 256 x 16B chunks each, 1 per thread per tile
    {
      int row = tid >> 2;
      int c   = (tid & 3) << 3;
      const unsigned short* gh = BHi + (size_t)(nBase + row) * LDB + k0 + c;
      const unsigned short* gl = BLo + (size_t)(nBase + row) * LDB + k0 + c;
      unsigned short* lh = &Bhs[buf][row * 32 + c];
      unsigned short* ll = &Bls[buf][row * 32 + c];
#if USE_ASYNC
      async_cp16(gh, lh);
      async_cp16(gl, ll);
#else
      *(uint4*)lh = *(const uint4*)gh;
      *(uint4*)ll = *(const uint4*)gl;
#endif
    }
  };

  v8f acc[2][2] = {};
  stage(0, 0);  // prefetch first tile

  for (int k0 = 0, it = 0; k0 < K; k0 += 32, ++it) {
    const int cur = it & 1;
#if USE_ASYNC
    __builtin_amdgcn_s_wait_asynccnt(0);  // this wave's copies for `cur` landed
#endif
    __syncthreads();                       // everyone's copies landed; prev compute done
    if (k0 + 32 < K) stage(cur ^ 1, k0 + 32);  // prefetch next while we convert/compute

    // convert Araw[cur] f32 -> hi/lo bf16
#pragma unroll
    for (int r = 0; r < 4; ++r) {
      int idx = tid + r * 256;
      int row = idx >> 3;
      int c   = (idx & 7) << 2;
      float4 v = *(const float4*)&Araw[cur][row * 32 + c];
      float vv[4] = {v.x, v.y, v.z, v.w};
#pragma unroll
      for (int j = 0; j < 4; ++j) {
        unsigned short h, l;
        split_bf16_fast(vv[j], h, l);
        Ahi[row * 32 + c + j] = h;
        Alo[row * 32 + c + j] = l;
      }
    }
    __syncthreads();

    // fragments (contiguous k = 16*half + s in both A and B)
    v16bf ah[2], al[2], bh[2], bl[2];
#pragma unroll
    for (int i = 0; i < 2; ++i) {
      int m = wm * 32 + i * 16 + l16;
      bf16x8 h0 = *(const bf16x8*)(Ahi + m * 32 + 16 * half);
      bf16x8 h1 = *(const bf16x8*)(Ahi + m * 32 + 16 * half + 8);
      bf16x8 l0 = *(const bf16x8*)(Alo + m * 32 + 16 * half);
      bf16x8 l1 = *(const bf16x8*)(Alo + m * 32 + 16 * half + 8);
      ah[i] = __builtin_shufflevector(h0, h1, 0,1,2,3,4,5,6,7,8,9,10,11,12,13,14,15);
      al[i] = __builtin_shufflevector(l0, l1, 0,1,2,3,4,5,6,7,8,9,10,11,12,13,14,15);
    }
#pragma unroll
    for (int j = 0; j < 2; ++j) {
      int n = wn * 32 + j * 16 + l16;
      bh[j] = *(const v16bf*)(&Bhs[cur][n * 32 + 16 * half]);
      bl[j] = *(const v16bf*)(&Bls[cur][n * 32 + 16 * half]);
    }

    // bf16x3 compensated products
#pragma unroll
    for (int i = 0; i < 2; ++i) {
#pragma unroll
      for (int j = 0; j < 2; ++j) {
        acc[i][j] = __builtin_amdgcn_wmma_f32_16x16x32_bf16(
            false, ah[i], false, bh[j], (short)0, acc[i][j], false, false);
        acc[i][j] = __builtin_amdgcn_wmma_f32_16x16x32_bf16(
            false, ah[i], false, bl[j], (short)0, acc[i][j], false, false);
        acc[i][j] = __builtin_amdgcn_wmma_f32_16x16x32_bf16(
            false, al[i], false, bh[j], (short)0, acc[i][j], false, false);
      }
    }
  }

  // epilogue: y = x*mask1[d] + relu(x)*mask2[d]; C/D layout N=lane&15, M=r+8*(lane>>4)
#pragma unroll
  for (int i = 0; i < 2; ++i) {
#pragma unroll
    for (int j = 0; j < 2; ++j) {
      int nCol = nBase + wn * 32 + j * 16 + l16;
      float m1 = mask1[nCol];
      float m2 = mask2[nCol];
#pragma unroll
      for (int r = 0; r < 8; ++r) {
        int mRow = mBase + wm * 32 + i * 16 + half * 8 + r;
        float x = acc[i][j][r];
        C[(size_t)mRow * LDC + nCol] = x * m1 + fmaxf(x, 0.0f) * m2;
      }
    }
  }
}

extern "C" void kernel_launch(void* const* d_in, const int* in_sizes, int n_in,
                              void* d_out, int out_size, void* d_ws, size_t ws_size,
                              hipStream_t stream) {
  (void)in_sizes; (void)n_in; (void)out_size; (void)ws_size;
  const float* data       = (const float*)d_in[0];
  const float* base       = (const float*)d_in[1];
  const float* base_mask  = (const float*)d_in[2];
  const float* eye        = (const float*)d_in[3];
  const float* tdiag      = (const float*)d_in[4];
  const float* tdiag_mask = (const float*)d_in[5];
  const float* udiag      = (const float*)d_in[6];
  const float* mask1      = (const float*)d_in[7];
  const float* mask2      = (const float*)d_in[8];
  float* out = (float*)d_out;
  float* ws  = (float*)d_ws;

  const size_t MAT = (size_t)DIM * DIM;
  float* layer = ws + 0 * MAT;
  float* neg2  = ws + 1 * MAT;
  float* diagM = ws + 2 * MAT;
  float* T1    = ws + 3 * MAT;
  float* W     = ws + 4 * MAT;
  unsigned short* Whi = (unsigned short*)(ws + 5 * MAT);   // 2 MB
  unsigned short* Wlo = Whi + MAT;                          // 2 MB  (24 MB ws total)

  // 1) elementwise prep
  prep_kernel<<<dim3((DIM * DIM) / 256), 256, 0, stream>>>(
      base, base_mask, eye, tdiag, tdiag_mask, udiag, layer, neg2, diagM);

  // 2) T1 = diag @ (2I - layer)
  gemm_small<<<dim3(DIM / 64, DIM / 128), 256, 0, stream>>>(
      diagM, neg2, T1, DIM, DIM, DIM, DIM);

  // 3) W = layer @ T1   (row-major [d][k])
  gemm_small<<<dim3(DIM / 64, DIM / 128), 256, 0, stream>>>(
      layer, T1, W, DIM, DIM, DIM, DIM);

  // 4) split W once into bf16 hi/lo
  splitw_kernel<<<dim3((DIM * DIM) / 256), 256, 0, stream>>>(W, Whi, Wlo);

  // 5) out = eps( data @ W^T ), N-fastest grid for L2 reuse of data tiles
  main_gemm<<<dim3(DIM / 64, BATCH / 128), 256, 0, stream>>>(
      data, Whi, Wlo, out, mask1, mask2);
}
